// HNN_10488310137126
// MI455X (gfx1250) — matrix operand order repaired and checked by reading, not compiled, not run on hardware
//
#include <hip/hip_runtime.h>
#include <hip/hip_bf16.h>

typedef __attribute__((ext_vector_type(16))) _Float16 v16h;
typedef __attribute__((ext_vector_type(8)))  _Float16 v8h;
typedef __attribute__((ext_vector_type(8)))  float    v8f;
typedef __attribute__((ext_vector_type(8)))  int      v8i;

#define HID 128
#define DTC 0.02f

// sigma: swap bits 3 and 4 of a 0..127 feature index. Self-inverse.
// Physical B-fragment position k carries logical hidden feature sigma(k) when each
// lane packs its own D-fragment rows in place; all A matrices contracting over the
// hidden dimension are staged with sigma applied to their k columns to compensate.
__host__ __device__ __forceinline__ int sigma(int k) {
    return (k & ~24) | ((k & 8) << 1) | ((k & 16) >> 1);
}

__device__ __forceinline__ float fast_tanh(float x) {
#if __has_builtin(__builtin_amdgcn_tanhf)
    return __builtin_amdgcn_tanhf(x);
#else
    return tanhf(x);
#endif
}

__device__ __forceinline__ v8f wmma_f16(v16h a, v16h b, v8f c) {
    return __builtin_amdgcn_wmma_f32_16x16x32_f16(false, a, false, b, (short)0, c, false, false);
}

// Load one 16x32 f16 A-fragment from an LDS row-major matrix with given row stride.
// k order used here (contiguous, matched by the B operands): lanes 0-15 hold M=lane, K = {kb..kb+15} with kb = 32t
// (lanes 16-31 use kb = 32t+16). Both runs are 16B-contiguous -> ds_load_b128 each.
__device__ __forceinline__ v16h load_afrag(const _Float16* W, int stride, int m0, int t, int mlane, bool hi) {
    const _Float16* rp = W + (m0 + mlane) * stride + 32 * t + (hi ? 16 : 0);
    v8h lo = *(const v8h*)rp;
    v8h hv = *(const v8h*)(rp + 8);
    v16h a;
#pragma unroll
    for (int e = 0; e < 8; ++e) { a[e] = lo[e]; a[8 + e] = hv[e]; }
    return a;
}

// Two adjacent 16-row output blocks (b,b+1) x two sample tiles (Xa,Xb), K=128.
// Software-pipelined: k-step t+1 fragments are loaded before the k-step t WMMAs,
// so the 4 WMMAs cover the LDS latency of the next loads (no full dscnt drains).
__device__ __forceinline__ void gemm_quad(const _Float16* W, int b,
                                          const v16h* Xa, const v16h* Xb,
                                          v8f& a00, v8f& a01, v8f& a10, v8f& a11,
                                          int mlane, bool hi) {
    v16h f0 = load_afrag(W, HID, 16 * b,       0, mlane, hi);
    v16h f1 = load_afrag(W, HID, 16 * (b + 1), 0, mlane, hi);
#pragma unroll
    for (int t = 0; t < 4; ++t) {
        v16h c0 = f0, c1 = f1;
        if (t < 3) {
            f0 = load_afrag(W, HID, 16 * b,       t + 1, mlane, hi);
            f1 = load_afrag(W, HID, 16 * (b + 1), t + 1, mlane, hi);
        }
        a00 = wmma_f16(c0, Xa[t], a00);
        a10 = wmma_f16(c0, Xb[t], a10);
        a01 = wmma_f16(c1, Xa[t], a01);
        a11 = wmma_f16(c1, Xb[t], a11);
    }
}

// Pure per-lane pack of two f32 D-fragments (blocks 2t, 2t+1) into the f16 B-fragment
// for tile t. No cross-lane movement needed thanks to the sigma feature permutation.
__device__ __forceinline__ v16h pack2(v8f a0, v8f a1) {
    v8i o;
#pragma unroll
    for (int r = 0; r < 4; ++r) {
        o[r]     = __builtin_bit_cast(int, __builtin_amdgcn_cvt_pkrtz(a0[2 * r], a0[2 * r + 1]));
        o[4 + r] = __builtin_bit_cast(int, __builtin_amdgcn_cvt_pkrtz(a1[2 * r], a1[2 * r + 1]));
    }
    return __builtin_bit_cast(v16h, o);
}

__global__ __launch_bounds__(256) __attribute__((amdgpu_waves_per_eu(1)))
void hnn_grad_kernel(
    const float* __restrict__ q, const float* __restrict__ p, const float* __restrict__ fext,
    const float* __restrict__ W1, const float* __restrict__ b1,
    const float* __restrict__ W2, const float* __restrict__ b2,
    const float* __restrict__ W3, const float* __restrict__ b3,
    const float* __restrict__ W4,
    float* __restrict__ out, int nTiles, int nWaves)
{
    __shared__ __align__(32) _Float16 sW2[HID * HID];    // A for backward U1: [m][k] = W2[m][sigma(k)]
    __shared__ __align__(32) _Float16 sW3[HID * HID];    // A for backward U2: [m][k] = W3[m][sigma(k)]
    __shared__ __align__(32) _Float16 sW2T[HID * HID];   // A for fwd layer 2: [m][k] = W2[sigma(k)][m]
    __shared__ __align__(32) _Float16 sW3T[HID * HID];   // A for fwd layer 3: [m][k] = W3[sigma(k)][m]
    __shared__ __align__(32) _Float16 sW1A[HID * 32];    // layer-1 A: k=0,1 = W1 cols, k=2 = b1, rest 0
    __shared__ __align__(32) _Float16 sW1O[16 * HID];    // head A: rows 0,1 = W1[m][sigma(k)], rest 0
    __shared__ __align__(32) float sb2[HID];
    __shared__ __align__(32) float sb3[HID];
    __shared__ __align__(32) float sw4[HID];

    const int tid = threadIdx.x;

    // Stage weights: f32 global -> f16 LDS with the sigma column permutation.
    for (int idx = tid; idx < HID * HID; idx += 256) {
        int i = idx >> 7, j = idx & (HID - 1);
        int si = sigma(i), sj = sigma(j);
        _Float16 w2 = (_Float16)W2[idx];
        _Float16 w3 = (_Float16)W3[idx];
        sW2[i * HID + sj] = w2;   // sW2[m][k] = W2[m][sigma(k)]
        sW3[i * HID + sj] = w3;
        sW2T[j * HID + si] = w2;  // sW2T[m][k] = W2[sigma(k)][m]
        sW3T[j * HID + si] = w3;
    }
    for (int idx = tid; idx < HID * 32; idx += 256) {
        int m = idx >> 5, k = idx & 31;
        // layer-1 k is the (q,p,1) input: k=0 -> W1 row 0, k=1 -> W1 row 1, k=2 -> b1 (bias via
        // constant-1 input), rest zero. No sigma on this k.
        float v = (k < 2) ? W1[k * HID + m] : ((k == 2) ? b1[m] : 0.0f);
        sW1A[idx] = (_Float16)v;
    }
    for (int idx = tid; idx < 16 * HID; idx += 256) {
        int m = idx >> 7, k = idx & (HID - 1);
        sW1O[m * HID + sigma(k)] = (_Float16)((m < 2) ? W1[m * HID + k] : 0.0f);
    }
    if (tid < HID) {
        sb2[tid] = b2[tid];  sb3[tid] = b3[tid];
        sw4[tid] = W4[tid];
    }
    __syncthreads();

    const int  lane  = tid & 31;
    const bool hi    = (lane & 16) != 0;
    const int  nl    = lane & 15;
    const int  gwave = blockIdx.x * 8 + (tid >> 5);

    // Runtime-zero offset "redefined" every iteration: keeps all LDS addresses
    // loop-variant so LICM cannot hoist weight fragments out of the loop.
    unsigned fudge = 0;

    for (int tile = gwave; tile < nTiles; tile += nWaves) {
        asm volatile("" : "+v"(fudge));
        const _Float16* w2p  = sW2  + fudge;
        const _Float16* w3p  = sW3  + fudge;
        const _Float16* w2tp = sW2T + fudge;
        const _Float16* w3tp = sW3T + fudge;
        const _Float16* w1Ap = sW1A + fudge;
        const _Float16* w1Op = sW1O + fudge;
        const float*    b2p  = sb2  + fudge;
        const float*    b3p  = sb3  + fudge;
        const float*    w4p  = sw4  + fudge;

        const int base = tile * 32;
        const int sA = base + nl;
        const int sB = base + 16 + nl;
        const float qa = q[sA], pa = p[sA];
        const float qb = q[sB], pb = p[sB];

        // ---- Layer 1 as WMMA: z1 = W1^T s + b1, K padded to 32 (k=2 carries the bias).
        v16h X1a[4], X1b[4];
        {
            const _Float16 one = (_Float16)1.0f, zero = (_Float16)0.0f;
            v16h Sa = {}, Sb = {};
            Sa[0] = hi ? zero : (_Float16)qa;
            Sa[1] = hi ? zero : (_Float16)pa;
            Sa[2] = hi ? zero : one;           // constant-1 input -> adds b1 column
            Sb[0] = hi ? zero : (_Float16)qb;
            Sb[1] = hi ? zero : (_Float16)pb;
            Sb[2] = hi ? zero : one;
#pragma unroll
            for (int bb = 0; bb < 4; ++bb) {
                int b = 2 * bb;
                v8f a00 = {}, a01 = {}, a10 = {}, a11 = {};
                v16h f0 = load_afrag(w1Ap, 32, 16 * b,       0, nl, hi);
                v16h f1 = load_afrag(w1Ap, 32, 16 * (b + 1), 0, nl, hi);
                a00 = wmma_f16(f0, Sa, a00);  a10 = wmma_f16(f0, Sb, a10);
                a01 = wmma_f16(f1, Sa, a01);  a11 = wmma_f16(f1, Sb, a11);
                v8f t00, t01, t10, t11;
#pragma unroll
                for (int r = 0; r < 8; ++r) {
                    t00[r] = fast_tanh(a00[r]);  t01[r] = fast_tanh(a01[r]);
                    t10[r] = fast_tanh(a10[r]);  t11[r] = fast_tanh(a11[r]);
                }
                X1a[bb] = pack2(t00, t01);
                X1b[bb] = pack2(t10, t11);
            }
        }

        // ---- Layer 2 forward: X2 = tanh(W2^T X1 + b2)
        v16h X2a[4], X2b[4];
#pragma unroll
        for (int bb = 0; bb < 4; ++bb) {
            int b = 2 * bb;
            v8f bias0 = *(const v8f*)&b2p[16 * b       + (hi ? 8 : 0)];
            v8f bias1 = *(const v8f*)&b2p[16 * (b + 1) + (hi ? 8 : 0)];
            v8f a00 = bias0, a01 = bias1, a10 = bias0, a11 = bias1;
            gemm_quad(w2tp, b, X1a, X1b, a00, a01, a10, a11, nl, hi);
            v8f t00, t01, t10, t11;
#pragma unroll
            for (int r = 0; r < 8; ++r) {
                t00[r] = fast_tanh(a00[r]);  t01[r] = fast_tanh(a01[r]);
                t10[r] = fast_tanh(a10[r]);  t11[r] = fast_tanh(a11[r]);
            }
            X2a[bb] = pack2(t00, t01);
            X2b[bb] = pack2(t10, t11);
        }

        // ---- Layer 3 fused with head gradient: G3 = W4 .* (1 - tanh(z3)^2)
        v16h G3a[4], G3b[4];
#pragma unroll
        for (int bb = 0; bb < 4; ++bb) {
            int b = 2 * bb;
            v8f bias0 = *(const v8f*)&b3p[16 * b       + (hi ? 8 : 0)];
            v8f bias1 = *(const v8f*)&b3p[16 * (b + 1) + (hi ? 8 : 0)];
            v8f a00 = bias0, a01 = bias1, a10 = bias0, a11 = bias1;
            gemm_quad(w3tp, b, X2a, X2b, a00, a01, a10, a11, nl, hi);
            v8f w0 = *(const v8f*)&w4p[16 * b       + (hi ? 8 : 0)];
            v8f w1 = *(const v8f*)&w4p[16 * (b + 1) + (hi ? 8 : 0)];
            v8f g00, g01, g10, g11;
#pragma unroll
            for (int r = 0; r < 8; ++r) {
                float u0 = fast_tanh(a00[r]);  g00[r] = w0[r] * (1.0f - u0 * u0);
                float u1 = fast_tanh(a01[r]);  g01[r] = w1[r] * (1.0f - u1 * u1);
                float u2 = fast_tanh(a10[r]);  g10[r] = w0[r] * (1.0f - u2 * u2);
                float u3 = fast_tanh(a11[r]);  g11[r] = w1[r] * (1.0f - u3 * u3);
            }
            G3a[bb] = pack2(g00, g01);
            G3b[bb] = pack2(g10, g11);
        }

        // ---- Backward: G2 = (W3 G3) .* (1 - X2^2)   (positions align: same packing)
        v16h G2a[4], G2b[4];
#pragma unroll
        for (int bb = 0; bb < 4; ++bb) {
            int b = 2 * bb;
            v8f a00 = {}, a01 = {}, a10 = {}, a11 = {};
            gemm_quad(w3p, b, G3a, G3b, a00, a01, a10, a11, nl, hi);
            v16h ua = pack2(a00, a01);
            v16h ub = pack2(a10, a11);
            v16h xa = X2a[bb], xb = X2b[bb];
            G2a[bb] = ua - ua * (xa * xa);   // packed f16 (v_pk_*)
            G2b[bb] = ub - ub * (xb * xb);
        }

        // ---- Backward: G1 = (W2 G2) .* (1 - X1^2)
        v16h G1a[4], G1b[4];
#pragma unroll
        for (int bb = 0; bb < 4; ++bb) {
            int b = 2 * bb;
            v8f a00 = {}, a01 = {}, a10 = {}, a11 = {};
            gemm_quad(w2p, b, G2a, G2b, a00, a01, a10, a11, nl, hi);
            v16h ua = pack2(a00, a01);
            v16h ub = pack2(a10, a11);
            v16h xa = X1a[bb], xb = X1b[bb];
            G1a[bb] = ua - ua * (xa * xa);
            G1b[bb] = ub - ub * (xb * xb);
        }

        // ---- Head as WMMA: dH/ds = W1 G1; rows 0/1 of the D block hold dH/dq, dH/dp.
        v8f ha = {}, hb = {};
        {
            v16h f = load_afrag(w1Op, HID, 0, 0, nl, hi);
#pragma unroll
            for (int t = 0; t < 4; ++t) {
                v16h c = f;
                if (t < 3) f = load_afrag(w1Op, HID, 0, t + 1, nl, hi);
                ha = wmma_f16(c, G1a[t], ha);
                hb = wmma_f16(c, G1b[t], hb);
            }
        }

        if (!hi) {   // lanes 0-15: row0 = dH/dq, row1 = dH/dp, column = sample
            float2 oa;
            oa.x = DTC * ha[1];                  // dq_dt = dH/dp
            oa.y = DTC * (-ha[0] + fext[sA]);    // dp_dt = -dH/dq + F_ext
            *(float2*)(out + 2 * sA) = oa;
            float2 ob;
            ob.x = DTC * hb[1];
            ob.y = DTC * (-hb[0] + fext[sB]);
            *(float2*)(out + 2 * sB) = ob;
        }
    }
}

extern "C" void kernel_launch(void* const* d_in, const int* in_sizes, int n_in,
                              void* d_out, int out_size, void* d_ws, size_t ws_size,
                              hipStream_t stream) {
    const float* q    = (const float*)d_in[0];
    const float* p    = (const float*)d_in[1];
    const float* fext = (const float*)d_in[2];
    const float* W1   = (const float*)d_in[3];
    const float* b1   = (const float*)d_in[4];
    const float* W2   = (const float*)d_in[5];
    const float* b2   = (const float*)d_in[6];
    const float* W3   = (const float*)d_in[7];
    const float* b3   = (const float*)d_in[8];
    const float* W4   = (const float*)d_in[9];
    // d_in[10] = b4: not needed for the gradient.

    const int Bn     = in_sizes[0];
    const int nTiles = Bn / 32;           // 32 samples per wave iteration
    const int blocks = 512;               // x8 waves -> 4096 wave "streams"
    const int nWaves = blocks * 8;

    hnn_grad_kernel<<<blocks, 256, 0, stream>>>(q, p, fext, W1, b1, W2, b2, W3, b3, W4,
                                                (float*)d_out, nTiles, nWaves);
}
